// MLPWrapper_31671088841243
// MI455X (gfx1250) — compile-verified, matrix operand order checked by reading, not run on hardware
//
#include <hip/hip_runtime.h>

// ---------------------------------------------------------------------------
// CDNA5 (gfx1250) fused SwiGLU MLP, bf16 WMMA path.
//  - all weights pre-converted f32->bf16 once (halves B traffic; 128MB
//    Hadamard matrices become L2-resident on the 192MB L2)
//  - straight-copy tiles use GLOBAL_LOAD_ASYNC_TO_LDS_B128 (ASYNCcnt path)
//  - 128x128 C-tile / block, 8 wave32s, 2x4 WMMA accum per wave, K-block 32
// ---------------------------------------------------------------------------

typedef __attribute__((ext_vector_type(16))) __bf16 v16bf;
typedef __attribute__((ext_vector_type(8)))  float  v8f;

#define TILE_M 128
#define TILE_N 128
#define TILE_K 32

// async global->LDS copy of 16 bytes (8 bf16) per lane, tracked by ASYNCcnt
__device__ __forceinline__ void async_copy_16B(const void* gsrc, void* ldst)
{
    unsigned           lds = (unsigned)(unsigned long long)ldst;
    unsigned long long ga  = (unsigned long long)gsrc;
    asm volatile("global_load_async_to_lds_b128 %0, %1, off"
                 :: "v"(lds), "v"(ga) : "memory");
}

__device__ __forceinline__ void wait_asynccnt0()
{
    asm volatile("s_wait_asynccnt 0" ::: "memory");
}

// C[M,N] = A[M,K] * op(B).  TRANSB: B stored [N,K] row-major (K-contiguous).
// !TRANSB: B stored [K,N] row-major (N-contiguous).  B is bf16 (pre-converted).
// A is AT (float: convert on load; __bf16: async straight copy). C is CT.
template <typename AT, bool TRANSB, typename CT>
__global__ __launch_bounds__(256)
void gemm_wmma_bf16(const AT* __restrict__ A, const __bf16* __restrict__ B,
                    CT* __restrict__ C, int M, int N, int K,
                    int lda, int ldb, int ldc)
{
    __shared__ alignas(64) __bf16 As[TILE_M][TILE_K];   // [m][k] 8 KB
    __shared__ alignas(64) __bf16 Bs[TILE_N][TILE_K];   // [n][k] 8 KB

    const int tid  = threadIdx.x;
    const int lane = tid & 31;
    const int wid  = tid >> 5;              // 8 waves
    const int m0   = blockIdx.y * TILE_M;
    const int n0   = blockIdx.x * TILE_N;

    const int mwave = (wid & 3) * 32;       // 4 waves down M
    const int nwave = (wid >> 2) * 64;      // 2 waves across N

    v8f acc[2][4];
#pragma unroll
    for (int mi = 0; mi < 2; ++mi)
#pragma unroll
        for (int ni = 0; ni < 4; ++ni)
            acc[mi][ni] = (v8f){0.f,0.f,0.f,0.f,0.f,0.f,0.f,0.f};

    // tile-loader mapping: 256 threads, each owns 16 contiguous K elements
    const int arow  = tid >> 1;             // 0..127
    const int ahalf = (tid & 1) * 16;       // 0 or 16

    constexpr bool ASYNC_A = (sizeof(AT) == 2);
    constexpr bool ASYNC_B = TRANSB;

    for (int k0 = 0; k0 < K; k0 += TILE_K) {
        // ---- A tile ----
        if constexpr (ASYNC_A) {
            const AT* g = &A[(size_t)(m0 + arow) * lda + (k0 + ahalf)];
            async_copy_16B(g,     &As[arow][ahalf]);
            async_copy_16B(g + 8, &As[arow][ahalf + 8]);
        } else {
            if (k0 + TILE_K < K)
                __builtin_prefetch(&A[(size_t)(m0 + arow) * lda + (k0 + TILE_K + ahalf)], 0, 1);
#pragma unroll
            for (int i = 0; i < 16; ++i)
                As[arow][ahalf + i] = (__bf16)A[(size_t)(m0 + arow) * lda + (k0 + ahalf + i)];
        }

        // ---- B tile into [n][k] LDS layout ----
        if constexpr (TRANSB) {
            // [N,K] row-major, K-contiguous: straight async copy
            const __bf16* g = &B[(size_t)(n0 + arow) * ldb + (k0 + ahalf)];
            async_copy_16B(g,     &Bs[arow][ahalf]);
            async_copy_16B(g + 8, &Bs[arow][ahalf + 8]);
        } else {
            // [K,N] row-major, N-contiguous: transpose into LDS
            const int bk   = tid >> 3;          // k in tile (0..31)
            const int bseg = (tid & 7) * 16;    // n offset
            if (k0 + TILE_K < K)
                __builtin_prefetch(&B[(size_t)(k0 + TILE_K + bk) * ldb + (n0 + bseg)], 0, 1);
#pragma unroll
            for (int i = 0; i < 16; ++i)
                Bs[bseg + i][bk] = B[(size_t)(k0 + bk) * ldb + (n0 + bseg + i)];
        }

        if constexpr (ASYNC_A || ASYNC_B)
            wait_asynccnt0();
        __syncthreads();

        // ---- per-wave fragments: contiguous 32B LDS reads ----
        v16bf afrag[2], bfrag[4];
#pragma unroll
        for (int mi = 0; mi < 2; ++mi) {
            const int row = mwave + mi * 16 + (lane & 15);
            afrag[mi] = *reinterpret_cast<const v16bf*>(&As[row][(lane >> 4) * 16]);
        }
#pragma unroll
        for (int ni = 0; ni < 4; ++ni) {
            const int col = nwave + ni * 16 + (lane & 15);
            bfrag[ni] = *reinterpret_cast<const v16bf*>(&Bs[col][(lane >> 4) * 16]);
        }

        // ---- 8 x v_wmma_f32_16x16x32_bf16 ----
#pragma unroll
        for (int mi = 0; mi < 2; ++mi)
#pragma unroll
            for (int ni = 0; ni < 4; ++ni)
                acc[mi][ni] = __builtin_amdgcn_wmma_f32_16x16x32_bf16(
                    false, afrag[mi], false, bfrag[ni],
                    (short)0, acc[mi][ni], false, false);

        __syncthreads();
    }

    // ---- store: VGPR r of a 16x16 tile -> (M = r + 8*(lane>=16), N = lane&15)
#pragma unroll
    for (int mi = 0; mi < 2; ++mi) {
#pragma unroll
        for (int ni = 0; ni < 4; ++ni) {
            const int row0 = m0 + mwave + mi * 16 + (lane >> 4) * 8;
            const int col  = n0 + nwave + ni * 16 + (lane & 15);
#pragma unroll
            for (int r = 0; r < 8; ++r)
                C[(size_t)(row0 + r) * ldc + col] = (CT)acc[mi][ni][r];
        }
    }
}

// f32 -> bf16 weight conversion, 8 elements / thread (n divisible by 2048)
__global__ __launch_bounds__(256)
void cvt_f32_bf16(const float* __restrict__ src, __bf16* __restrict__ dst, size_t n)
{
    size_t i = ((size_t)blockIdx.x * blockDim.x + threadIdx.x) * 8;
    if (i + 8 <= n) {
        float4 a = *reinterpret_cast<const float4*>(src + i);
        float4 b = *reinterpret_cast<const float4*>(src + i + 4);
        dst[i + 0] = (__bf16)a.x;  dst[i + 1] = (__bf16)a.y;
        dst[i + 2] = (__bf16)a.z;  dst[i + 3] = (__bf16)a.w;
        dst[i + 4] = (__bf16)b.x;  dst[i + 5] = (__bf16)b.y;
        dst[i + 6] = (__bf16)b.z;  dst[i + 7] = (__bf16)b.w;
    }
}

// gated = up * silu(gate), bf16 in/out, f32 math
__global__ __launch_bounds__(256)
void swiglu_kernel(const __bf16* __restrict__ up, const __bf16* __restrict__ gate,
                   __bf16* __restrict__ out, size_t n)
{
    size_t i = (size_t)blockIdx.x * blockDim.x + threadIdx.x;
    if (i < n) {
        float u = (float)up[i];
        float g = (float)gate[i];
        float s = g / (1.0f + __expf(-g));
        out[i] = (__bf16)(u * s);
    }
}

extern "C" void kernel_launch(void* const* d_in, const int* in_sizes, int n_in,
                              void* d_out, int out_size, void* d_ws, size_t ws_size,
                              hipStream_t stream)
{
    (void)in_sizes; (void)n_in; (void)out_size; (void)ws_size;

    const float* x        = (const float*)d_in[0];   // [M,D]
    const float* w_up     = (const float*)d_in[1];   // [F,D]
    const float* w_gate   = (const float*)d_in[2];   // [F,D]
    const float* w_down   = (const float*)d_in[3];   // [D,F]
    const float* h_up_T   = (const float*)d_in[4];   // [F,F]
    const float* h_gate_T = (const float*)d_in[5];   // [F,F]
    const float* h_down   = (const float*)d_in[6];   // [F,F]
    float*       out      = (float*)d_out;           // [M,D]

    const int M = 4 * 2048;   // B*S
    const int D = 2048;
    const int F = 8192;

    // workspace layout (all bf16):
    //   U0..U3 : 4 x M*F activation buffers
    //   wbu, wbg : F*D     wbd : D*F     hbu, hbg, hbd : F*F
    const size_t bufMF = (size_t)M * F;
    const size_t bufFD = (size_t)F * D;
    const size_t bufFF = (size_t)F * F;

    __bf16* U0  = (__bf16*)d_ws;
    __bf16* U1  = U0 + bufMF;
    __bf16* U2  = U1 + bufMF;
    __bf16* U3  = U2 + bufMF;
    __bf16* wbu = U3 + bufMF;
    __bf16* wbg = wbu + bufFD;
    __bf16* wbd = wbg + bufFD;
    __bf16* hbu = wbd + bufFD;
    __bf16* hbg = hbu + bufFF;
    __bf16* hbd = hbg + bufFF;

    dim3 blk(256);

    // ---- one-time weight conversion f32 -> bf16 ----
    cvt_f32_bf16<<<dim3((unsigned)(bufFD / 2048)), blk, 0, stream>>>(w_up,     wbu, bufFD);
    cvt_f32_bf16<<<dim3((unsigned)(bufFD / 2048)), blk, 0, stream>>>(w_gate,   wbg, bufFD);
    cvt_f32_bf16<<<dim3((unsigned)(bufFD / 2048)), blk, 0, stream>>>(w_down,   wbd, bufFD);
    cvt_f32_bf16<<<dim3((unsigned)(bufFF / 2048)), blk, 0, stream>>>(h_up_T,   hbu, bufFF);
    cvt_f32_bf16<<<dim3((unsigned)(bufFF / 2048)), blk, 0, stream>>>(h_gate_T, hbg, bufFF);
    cvt_f32_bf16<<<dim3((unsigned)(bufFF / 2048)), blk, 0, stream>>>(h_down,   hbd, bufFF);

    dim3 gFF(F / TILE_N, M / TILE_M);   // 64 x 64
    dim3 gFD(D / TILE_N, M / TILE_M);   // 16 x 64

    // 1) up_raw = x @ w_up^T        (A f32, B bf16 NT)
    gemm_wmma_bf16<float, true, __bf16><<<gFF, blk, 0, stream>>>(
        x, wbu, U0, M, F, D, D, D, F);
    // 2) gate_raw = x @ w_gate^T
    gemm_wmma_bf16<float, true, __bf16><<<gFF, blk, 0, stream>>>(
        x, wbg, U1, M, F, D, D, D, F);
    // 3) up_rot = up_raw @ h_up_T   (A bf16 async, B bf16 NN)
    gemm_wmma_bf16<__bf16, false, __bf16><<<gFF, blk, 0, stream>>>(
        U0, hbu, U2, M, F, F, F, F, F);
    // 4) gate_rot = gate_raw @ h_gate_T
    gemm_wmma_bf16<__bf16, false, __bf16><<<gFF, blk, 0, stream>>>(
        U1, hbg, U3, M, F, F, F, F, F);
    // 5) gated = up_rot * silu(gate_rot)  -> U0
    {
        size_t n = (size_t)M * F;
        swiglu_kernel<<<dim3((unsigned)(n / 256)), blk, 0, stream>>>(U2, U3, U0, n);
    }
    // 6) gated_rot = gated @ h_down (NN) -> U1
    gemm_wmma_bf16<__bf16, false, __bf16><<<gFF, blk, 0, stream>>>(
        U0, hbd, U1, M, F, F, F, F, F);
    // 7) out = gated_rot @ w_down^T (NT, C f32)
    gemm_wmma_bf16<__bf16, true, float><<<gFD, blk, 0, stream>>>(
        U1, wbd, out, M, D, F, F, F, D);
}
